// NaiveSubMConv3d_45681272160694
// MI455X (gfx1250) — matrix operand order repaired and checked by reading, not compiled, not run on hardware
//
#include <hip/hip_runtime.h>

#define GRID_  128
#define BASE_  130            // GRID + 2
#define CIN    256
#define COUT   256
#define KVOL   27
#define HASH_BITS 19
#define HASH_SIZE (1 << HASH_BITS)
#define HASH_MASK (HASH_SIZE - 1)
#define MT     128            // output rows per block
#define LDS_STRIDE 264        // halves per LDS row (528 B = 33*16) -> conflict-free b128 reads

typedef __attribute__((ext_vector_type(16))) __bf16 v16bf;
typedef __attribute__((ext_vector_type(8)))  float  v8f;

union Frag16 { v16bf v; uint4 u[2]; };

__device__ __forceinline__ unsigned short f2bf(float f) {
  unsigned u = __float_as_uint(f);
  u += 0x7FFFu + ((u >> 16) & 1u);           // round-to-nearest-even
  return (unsigned short)(u >> 16);
}

__device__ __forceinline__ int pack_key(int b, int x, int y, int z) {
  return ((b * BASE_ + (x + 1)) * BASE_ + (y + 1)) * BASE_ + (z + 1);
}

__device__ __forceinline__ unsigned hash_key(int k) {
  unsigned h = (unsigned)k * 2654435761u;
  return (h >> (32 - HASH_BITS)) & HASH_MASK;
}

// ---------------- prep kernels ----------------

__global__ void fill_hash_k(int* __restrict__ tabk) {
  int i = blockIdx.x * blockDim.x + threadIdx.x;
  if (i < HASH_SIZE) tabk[i] = -1;
}

__global__ void fill_zero_k(unsigned* __restrict__ p, int n) {
  int i = blockIdx.x * blockDim.x + threadIdx.x;
  if (i < n) p[i] = 0u;
}

__global__ void cvt_x_k(const float* __restrict__ x, unsigned short* __restrict__ xb,
                        long long n4) {
  long long i = (long long)blockIdx.x * blockDim.x + threadIdx.x;
  if (i >= n4) return;
  float4 f = ((const float4*)x)[i];
  ushort4 o;
  o.x = f2bf(f.x); o.y = f2bf(f.y); o.z = f2bf(f.z); o.w = f2bf(f.w);
  ((ushort4*)xb)[i] = o;
}

// wT[o][n][k] = bf16(w[o][k][n])  -> per-lane contiguous K for B fragments
__global__ void cvt_w_k(const float* __restrict__ w, unsigned short* __restrict__ wT) {
  int i = blockIdx.x * blockDim.x + threadIdx.x;
  if (i >= KVOL * CIN * COUT) return;
  int o = i >> 16;
  int rem = i & 65535;
  int n = rem >> 8;
  int k = rem & 255;
  wT[i] = f2bf(w[(o << 16) + (k << 8) + n]);
}

__global__ void hash_insert_k(const int* __restrict__ coords, const int* __restrict__ bidx,
                              int* __restrict__ tabk, int* __restrict__ tabv, int N) {
  int i = blockIdx.x * blockDim.x + threadIdx.x;
  if (i >= N) return;
  int key = pack_key(bidx[i], coords[3 * i], coords[3 * i + 1], coords[3 * i + 2]);
  unsigned h = hash_key(key);
  for (;;) {
    int prev = atomicCAS(&tabk[h], -1, key);
    if (prev == -1) { tabv[h] = i; break; }
    h = (h + 1) & HASH_MASK;
  }
}

__global__ void build_nbr_k(const int* __restrict__ coords, const int* __restrict__ bidx,
                            const int* __restrict__ tabk, const int* __restrict__ tabv,
                            int* __restrict__ nbr, int N) {
  int i = blockIdx.x * blockDim.x + threadIdx.x;
  if (i >= N) return;
  int key = pack_key(bidx[i], coords[3 * i], coords[3 * i + 1], coords[3 * i + 2]);
  for (int o = 0; o < KVOL; ++o) {
    int ox = o / 9 - 1, oy = (o / 3) % 3 - 1, oz = o % 3 - 1;
    int tk = key + (ox * BASE_ + oy) * BASE_ + oz;
    unsigned h = hash_key(tk);
    int ni = -1;
    for (;;) {
      int k = tabk[h];
      if (k == tk) { ni = tabv[h]; break; }
      if (k == -1) break;
      h = (h + 1) & HASH_MASK;
    }
    nbr[(size_t)o * N + i] = ni;
  }
}

// ---------------- async staging helpers ----------------

__device__ __forceinline__ void wait_async_16() {
#if __has_builtin(__builtin_amdgcn_s_wait_asynccnt)
  __builtin_amdgcn_s_wait_asynccnt(16);
#else
  asm volatile("s_wait_asynccnt 0x10" ::: "memory");
#endif
}
__device__ __forceinline__ void wait_async_0() {
#if __has_builtin(__builtin_amdgcn_s_wait_asynccnt)
  __builtin_amdgcn_s_wait_asynccnt(0);
#else
  asm volatile("s_wait_asynccnt 0x0" ::: "memory");
#endif
}

// Stage one offset's 128 gathered rows (bf16) into LDS buffer at lds_base.
// thread t handles row t>>1, byte range [(t&1)*256, +256) of the 512B row.
// INST_OFFSET is added to BOTH the LDS and global address (ISA §10 async),
// so one base pair + 16 immediate-offset b128 issues covers 256 bytes.
__device__ __forceinline__ void stage_async(
    const unsigned short* __restrict__ xb, const unsigned short* __restrict__ zpage,
    const int* __restrict__ nbr, int o, int N, int r0, int row, int half,
    unsigned lds_base) {
  int grow = r0 + row;
  int ni = (grow < N) ? nbr[(size_t)o * N + grow] : -1;
  const unsigned short* src = (ni >= 0) ? (xb + (size_t)ni * CIN) : zpage;
  src += half * 128;                                            // +256 bytes
  unsigned lds_off = lds_base + (unsigned)row * (LDS_STRIDE * 2) + (unsigned)half * 256u;
  unsigned long long g = (unsigned long long)(uintptr_t)src;
#define ALD(IMM)                                                                 \
  asm volatile("global_load_async_to_lds_b128 %0, %1, off offset:" IMM           \
               :: "v"(lds_off), "v"(g) : "memory")
  ALD("0");   ALD("16");  ALD("32");  ALD("48");
  ALD("64");  ALD("80");  ALD("96");  ALD("112");
  ALD("128"); ALD("144"); ALD("160"); ALD("176");
  ALD("192"); ALD("208"); ALD("224"); ALD("240");
#undef ALD
}

// ---------------- main WMMA conv kernel ----------------
// block = 256 threads = 8 waves; tile = 128 rows x 256 out-channels.
// wave w owns col tiles {2w, 2w+1}; all 8 row sub-tiles.
// Double-buffered LDS: async-prefetch offset o+1 while computing offset o.

__global__ __launch_bounds__(256) void subm_conv_wmma_k(
    const unsigned short* __restrict__ xb,   // (N, 256) bf16
    const unsigned short* __restrict__ wT,   // (27, 256n, 256k) bf16
    const int* __restrict__ nbr,             // (27, N)
    const unsigned short* __restrict__ zpage,// 512B of zeros
    const float* __restrict__ bias,          // (256)
    float* __restrict__ out,                 // (N, 256) f32
    int N) {
  __shared__ __align__(16) unsigned short As[2 * MT * LDS_STRIDE];  // 132 KB

  const int t = threadIdx.x;
  const int wid = t >> 5;
  const int lane = t & 31;
  const int lane15 = lane & 15;
  const int grp = lane >> 4;           // 0: lanes 0-15, 1: lanes 16-31
  const int r0 = blockIdx.x * MT;

  const int srow = t >> 1;             // 0..127
  const int shalf = t & 1;

  const unsigned asBase = (unsigned)(uintptr_t)(void*)As;   // LDS byte offset
  const unsigned bufBytes = (unsigned)(MT * LDS_STRIDE * 2);

  v8f acc[8][2];
#pragma unroll
  for (int rs = 0; rs < 8; ++rs)
#pragma unroll
    for (int ct = 0; ct < 2; ++ct)
      acc[rs][ct] = (v8f){0.f, 0.f, 0.f, 0.f, 0.f, 0.f, 0.f, 0.f};

  // prologue: stage offset 0 into buffer 0
  stage_async(xb, zpage, nbr, 0, N, r0, srow, shalf, asBase);

  for (int o = 0; o < KVOL; ++o) {
    if (o + 1 < KVOL) {
      // prefetch next offset into the other buffer (last read in the previous iter)
      stage_async(xb, zpage, nbr, o + 1, N, r0, srow, shalf,
                  asBase + (unsigned)((o + 1) & 1) * bufBytes);
      wait_async_16();     // oldest 16 (= this offset's stage) complete
    } else {
      wait_async_0();
    }
    __syncthreads();

    const unsigned short* buf = As + (size_t)(o & 1) * (MT * LDS_STRIDE);

#pragma unroll
    for (int kb = 0; kb < 8; ++kb) {
      Frag16 b[2];
#pragma unroll
      for (int ct = 0; ct < 2; ++ct) {
        // B 32x16 bf16: lane = n; lanes 0-15 hold K0-15, lanes 16-31 K16-31.
        int n = (wid * 2 + ct) * 16 + lane15;
        const unsigned short* q =
            wT + (((size_t)o * COUT + n) << 8) + kb * 32 + grp * 16;
        b[ct].u[0] = ((const uint4*)q)[0];
        b[ct].u[1] = ((const uint4*)q)[1];
      }
#pragma unroll
      for (int rs = 0; rs < 8; ++rs) {
        // A 16x32 bf16: lanes 0-15 hold K0-15;
        // lanes 16-31 hold K16-31.
        Frag16 a;
        const unsigned short* p =
            buf + (rs * 16 + lane15) * LDS_STRIDE + kb * 32 + grp * 16;
        a.u[0] = *(const uint4*)p;
        a.u[1] = *(const uint4*)(p + 8);    // +16 bytes
#pragma unroll
        for (int ct = 0; ct < 2; ++ct)
          acc[rs][ct] = __builtin_amdgcn_wmma_f32_16x16x32_bf16(
              false, a.v, false, b[ct].v, (short)0, acc[rs][ct], false, false);
      }
    }
    __syncthreads();
  }

  // ---- epilogue: bias + store (C/D: VGPR j -> M=j (+8 for grp 1), lane15 = N) ----
#pragma unroll
  for (int ct = 0; ct < 2; ++ct) {
    int n = (wid * 2 + ct) * 16 + lane15;
    float bv = bias[n];
#pragma unroll
    for (int rs = 0; rs < 8; ++rs) {
#pragma unroll
      for (int j = 0; j < 8; ++j) {
        int row = r0 + rs * 16 + grp * 8 + j;
        if (row < N) out[(size_t)row * COUT + n] = acc[rs][ct][j] + bv;
      }
    }
  }
}

// ---------------- launch ----------------

extern "C" void kernel_launch(void* const* d_in, const int* in_sizes, int n_in,
                              void* d_out, int out_size, void* d_ws, size_t ws_size,
                              hipStream_t stream) {
  const float* x      = (const float*)d_in[0];
  const int*   coords = (const int*)d_in[1];
  const int*   bidx   = (const int*)d_in[2];
  const float* w      = (const float*)d_in[3];
  const float* bias   = (const float*)d_in[4];
  float* out = (float*)d_out;
  const int N = in_sizes[2];

  char* ws = (char*)d_ws;
  size_t off = 0;
  auto take = [&](size_t bytes) {
    char* p = ws + off;
    off = (off + bytes + 255) & ~(size_t)255;
    return p;
  };
  unsigned short* xb    = (unsigned short*)take((size_t)N * CIN * 2);
  unsigned short* wT    = (unsigned short*)take((size_t)KVOL * CIN * COUT * 2);
  int* tabk             = (int*)take((size_t)HASH_SIZE * 4);
  int* tabv             = (int*)take((size_t)HASH_SIZE * 4);
  int* nbr              = (int*)take((size_t)KVOL * N * 4);
  unsigned short* zpage = (unsigned short*)take(512);
  (void)n_in; (void)out_size; (void)ws_size;

  fill_hash_k<<<(HASH_SIZE + 255) / 256, 256, 0, stream>>>(tabk);
  fill_zero_k<<<1, 128, 0, stream>>>((unsigned*)zpage, 128);

  long long n4 = (long long)N * CIN / 4;
  cvt_x_k<<<(unsigned)((n4 + 255) / 256), 256, 0, stream>>>(x, xb, n4);

  cvt_w_k<<<(KVOL * CIN * COUT + 255) / 256, 256, 0, stream>>>(w, wT);

  hash_insert_k<<<(N + 255) / 256, 256, 0, stream>>>(coords, bidx, tabk, tabv, N);

  build_nbr_k<<<(N + 255) / 256, 256, 0, stream>>>(coords, bidx, tabk, tabv, nbr, N);

  subm_conv_wmma_k<<<(N + MT - 1) / MT, 256, 0, stream>>>(xb, wT, nbr, zpage, bias, out, N);
}
